// EncodeBlock_2379411882664
// MI455X (gfx1250) — matrix operand order repaired and checked by reading, not compiled, not run on hardware
//
#include <hip/hip_runtime.h>

// ---------------------------------------------------------------------------
// CDNA5 (gfx1250) WMMA types / helpers
// ---------------------------------------------------------------------------
typedef __attribute__((ext_vector_type(16))) __bf16 v16bf;
typedef __attribute__((ext_vector_type(8)))  float  v8f;

__device__ __forceinline__ v8f wmma_bf16(v16bf a, v16bf b, v8f c) {
    // D = A(16x32 bf16) * B(32x16 bf16) + C(16x16 f32)
    return __builtin_amdgcn_wmma_f32_16x16x32_bf16(
        /*neg_a=*/false, a, /*neg_b=*/false, b,
        /*c_mod=*/(short)0, c, /*reuse_a=*/false, /*reuse_b=*/false);
}

// A fragment: 16x32 tile at (m0,k0) of row-major bf16 matrix (lda elements).
// lane l: row = m0 + (l&15); elems e -> k = k0 + ((l>=16)?16:0) + e.
__device__ __forceinline__ v16bf load_a_frag(const __bf16* A, int lda,
                                             int m0, int k0, int lane) {
    int m  = m0 + (lane & 15);
    int kb = k0 + ((lane >> 4) << 4);
    const __bf16* p = A + (long)m * lda + kb;
    v16bf a;
#pragma unroll
    for (int e = 0; e < 8; ++e) a[e] = p[e];
#pragma unroll
    for (int e = 0; e < 8; ++e) a[8 + e] = p[8 + e];
    return a;
}

// B fragment from a TRANSPOSED operand: B[k][n] = Xt[n][k], Xt row-major
// (N x K, ldx = K). Per-lane 16 contiguous bf16 = 32B -> b128 loads.
// lane l: col = n0 + (l&15); elems e -> k = k0 + ((l>=16)?16:0) + e.
__device__ __forceinline__ v16bf load_bt_frag(const __bf16* Xt, int ldx,
                                              int k0, int n0, int lane) {
    int n  = n0 + (lane & 15);
    int kb = k0 + ((lane >> 4) << 4);
    const __bf16* p = Xt + (long)n * ldx + kb;
    v16bf b;
#pragma unroll
    for (int e = 0; e < 16; ++e) b[e] = p[e];
    return b;
}

__device__ __forceinline__ float swishf(float v) {
    return v / (1.f + __expf(-v));
}

// ---------------------------------------------------------------------------
// Problem dims
// ---------------------------------------------------------------------------
#define BATCH 4
#define SEQ   2048
#define EMB   256
#define NHEAD 8
#define DHEAD 32
#define ROWS  (BATCH * SEQ)   // 8192

// ---------------------------------------------------------------------------
// Conversions fp32 -> bf16 (weights stored TRANSPOSED: Wt[n][k])
// ---------------------------------------------------------------------------
__global__ void convert_x_kernel(const float* __restrict__ x,
                                 __bf16* __restrict__ xbf, int n) {
    int i = blockIdx.x * 256 + threadIdx.x;
    if (i < n) xbf[i] = (__bf16)x[i];
}

__global__ void convert_w_kernel(const float* wq, const float* wk,
                                 const float* wv, const float* wg,
                                 const float* wo, const float* wgate,
                                 const float* wlin, const float* wout,
                                 __bf16* wqkvgT, __bf16* woT,
                                 __bf16* wglT, __bf16* woutT) {
    int i = blockIdx.x * 256 + threadIdx.x;   // 2048 virtual rows x 256 k
    int n = i >> 8;          // virtual output-column index 0..2047
    int k = i & 255;
    if (n < 1024) {
        int seg = n >> 8, nn = n & 255;
        const float* w = (seg == 0) ? wq : (seg == 1) ? wk : (seg == 2) ? wv : wg;
        wqkvgT[n * 256 + k] = (__bf16)w[k * 256 + nn];
    } else if (n < 1280) {
        int nn = n - 1024;
        woT[nn * 256 + k] = (__bf16)wo[k * 256 + nn];
    } else if (n < 1792) {
        int nn = n - 1280;   // 0..511: gate | lin
        const float* w = (nn < 256) ? wgate : wlin;
        wglT[nn * 256 + k] = (__bf16)w[k * 256 + (nn & 255)];
    } else {
        int nn = n - 1792;
        woutT[nn * 256 + k] = (__bf16)wout[k * 256 + nn];
    }
}

// ---------------------------------------------------------------------------
// Fused projection: [q|k|v|gate_pre] = x @ [wq|wk|wv|wg]   (M=8192,K=256,N=1024)
// One wave computes a 32x64 tile (2 A frags x 4 B frags -> 8 WMMA per k-step,
// fully unrolled K). q,k,v scattered to (b,h,s,d)-major bf16; gate -> swish f32.
// ---------------------------------------------------------------------------
__global__ __launch_bounds__(32) void proj_kernel(
    const __bf16* __restrict__ xbf, const __bf16* __restrict__ wcatT,
    __bf16* __restrict__ qbf, __bf16* __restrict__ kbf,
    __bf16* __restrict__ vbf, float* __restrict__ g) {
    int lane = threadIdx.x;
    int n0 = blockIdx.x * 64;
    int m0 = blockIdx.y * 32;
    v8f acc[2][4] = {};
#pragma unroll
    for (int k0 = 0; k0 < EMB; k0 += 32) {
        v16bf a0 = load_a_frag(xbf, EMB, m0,      k0, lane);
        v16bf a1 = load_a_frag(xbf, EMB, m0 + 16, k0, lane);
#pragma unroll
        for (int j = 0; j < 4; ++j) {
            v16bf b = load_bt_frag(wcatT, EMB, k0, n0 + j * 16, lane);
            acc[0][j] = wmma_bf16(a0, b, acc[0][j]);
            acc[1][j] = wmma_bf16(a1, b, acc[1][j]);
        }
    }
#pragma unroll
    for (int half = 0; half < 2; ++half) {
        int mb = m0 + half * 16 + ((lane >> 4) << 3);
#pragma unroll
        for (int j = 0; j < 4; ++j) {
            int n   = n0 + j * 16 + (lane & 15);
            int seg = n >> 8;                 // 0=q 1=k 2=v 3=gate
            int nn  = n & 255;
            int h   = nn >> 5, d = nn & 31;
#pragma unroll
            for (int e = 0; e < 8; ++e) {
                int m = mb + e;               // m = b*SEQ + s
                float val = acc[half][j][e];
                if (seg == 3) {
                    g[(long)m * EMB + nn] = swishf(val);
                } else {
                    int b = m >> 11, s = m & (SEQ - 1);
                    long idx = (((long)(b * NHEAD + h) * SEQ) + s) * DHEAD + d;
                    __bf16* dst = (seg == 0) ? qbf : (seg == 1) ? kbf : vbf;
                    dst[idx] = (__bf16)val;
                }
            }
        }
    }
}

// ---------------------------------------------------------------------------
// Streaming retention: y[b,s,h,:] = sum_m (q.k^T * kappa^|n-m|) v
// Block = 8 waves; each wave owns a 16-row query tile; K tile row-major and
// V tile TRANSPOSED in LDS so every WMMA fragment read is contiguous.
// Decay applied in f32 accumulator; scores converted C-layout -> A-layout
// through padded LDS. 4 WMMAs per 32-key step. Next tile prefetched.
// ---------------------------------------------------------------------------
__global__ __launch_bounds__(256) void retention_kernel(
    const __bf16* __restrict__ qbf, const __bf16* __restrict__ kbf,
    const __bf16* __restrict__ vbf, float* __restrict__ y) {
    __shared__ __bf16 sK[32 * 32];     // [key][d]
    __shared__ __bf16 sVt[32 * 32];    // [d][key]  (transposed)
    __shared__ float  sS[8][16 * 33];  // per-wave score staging, padded

    int lane = threadIdx.x & 31;
    int wave = threadIdx.x >> 5;
    int bh   = blockIdx.x;              // b*NHEAD + h
    int h    = bh & (NHEAD - 1);
    int n0   = blockIdx.y * 128 + wave * 16;

    const __bf16* Q  = qbf + (long)bh * SEQ * DHEAD;
    const __bf16* Kp = kbf + (long)bh * SEQ * DHEAD;
    const __bf16* Vp = vbf + (long)bh * SEQ * DHEAD;

    // RetNet decay: kappa_h = (1 - exp(linspace(log 1/32, log 1/512, 8)[h])) * 0.8
    float l0  = -logf(32.f);
    float stp = (-logf(512.f) + logf(32.f)) / 7.f;
    float kap = (1.f - __expf(l0 + (float)h * stp)) * 0.8f;
    float lk  = logf(kap);

    v16bf aq = load_a_frag(Q, DHEAD, n0, 0, lane);   // 16x32 query tile
    v8f acc0 = {}, acc1 = {};
    float* sSw = sS[wave];

    int row  = threadIdx.x >> 2;        // 0..63 (32 K rows + 32 V rows)
    int quad = (threadIdx.x & 3) * 8;   // element offset within row

    for (int m0 = 0; m0 < SEQ; m0 += 32) {
        // cooperative K/V tile load: 16B per thread; V written transposed
        if (row < 32) {
            *(uint4*)(sK + row * 32 + quad) =
                *(const uint4*)(Kp + (long)(m0 + row) * DHEAD + quad);
        } else {
            int r = row - 32;
            uint4 tmp = *(const uint4*)(Vp + (long)(m0 + r) * DHEAD + quad);
            const __bf16* t = (const __bf16*)&tmp;
#pragma unroll
            for (int e = 0; e < 8; ++e) sVt[(quad + e) * 32 + r] = t[e];
        }
        // prefetch next tile while this step computes
        if (m0 + 32 < SEQ) {
            const __bf16* nsrc = (row < 32)
                ? (Kp + (long)(m0 + 32 + row) * DHEAD + quad)
                : (Vp + (long)(m0 + row) * DHEAD + quad);
            __builtin_prefetch(nsrc, 0, 3);
        }
        __syncthreads();

        // scores: two 16x16 tiles along the 32-key step
#pragma unroll
        for (int t = 0; t < 2; ++t) {
            v16bf bk = load_bt_frag(sK, 32, 0, t * 16, lane);  // K^T fragment
            v8f sc = wmma_bf16(aq, bk, (v8f){});
            int kg   = m0 + t * 16 + (lane & 15);
            int row0 = (lane >> 4) << 3;
#pragma unroll
            for (int e = 0; e < 8; ++e) {
                int qg = n0 + row0 + e;
                float dist = fabsf((float)(qg - kg));
                sSw[(row0 + e) * 33 + t * 16 + (lane & 15)] =
                    sc[e] * __expf(lk * dist);
            }
        }

        // scores (C-layout f32) -> A-layout bf16 fragment via LDS
        v16bf as;
        {
            int m  = lane & 15;
            int kb = (lane >> 4) << 4;
            const float* p = sSw + m * 33 + kb;
#pragma unroll
            for (int e = 0; e < 8; ++e) as[e] = (__bf16)p[e];
#pragma unroll
            for (int e = 0; e < 8; ++e) as[8 + e] = (__bf16)p[8 + e];
        }

        // y += S(16x32) * V(32x32): two 16x16 output tiles over DH.
        // sVt is [d][key] so these are load_bt-style contiguous reads.
        {
            int kb = (lane >> 4) << 4;   // key base
            const __bf16* p0 = sVt + (lane & 15) * 32 + kb;
            const __bf16* p1 = sVt + (16 + (lane & 15)) * 32 + kb;
            v16bf bv0, bv1;
#pragma unroll
            for (int e = 0; e < 16; ++e) { bv0[e] = p0[e]; bv1[e] = p1[e]; }
            acc0 = wmma_bf16(as, bv0, acc0);
            acc1 = wmma_bf16(as, bv1, acc1);
        }
        __syncthreads();
    }

    // write y[b][s][h][d] (f32)
    int b    = bh >> 3;
    int d    = lane & 15;
    int row0 = (lane >> 4) << 3;
#pragma unroll
    for (int e = 0; e < 8; ++e) {
        int s = n0 + row0 + e;
        long base = (((long)(b * SEQ + s) * NHEAD) + h) * DHEAD;
        y[base + d]      = acc0[e];
        y[base + 16 + d] = acc1[e];
    }
}

// ---------------------------------------------------------------------------
// Per-head GroupNorm (groups == heads, DH=32 == wave32) + swish gate
// ---------------------------------------------------------------------------
__global__ __launch_bounds__(256) void gn_gate_kernel(
    const float* __restrict__ y, const float* __restrict__ g,
    const float* __restrict__ gn_scale, const float* __restrict__ gn_bias,
    __bf16* __restrict__ z) {
    int row = blockIdx.x;
    int e   = threadIdx.x;            // e = h*32 + d; one head per wave32
    float v = y[(long)row * EMB + e];
    float mu = v;
#pragma unroll
    for (int off = 16; off > 0; off >>= 1) mu += __shfl_xor(mu, off);
    mu *= (1.f / 32.f);
    float dv  = v - mu;
    float var = dv * dv;
#pragma unroll
    for (int off = 16; off > 0; off >>= 1) var += __shfl_xor(var, off);
    var *= (1.f / 32.f);
    float yn = dv * rsqrtf(var + 1e-6f) * gn_scale[e] + gn_bias[e];
    z[(long)row * EMB + e] = (__bf16)(g[(long)row * EMB + e] * yn);
}

// ---------------------------------------------------------------------------
// Generic bf16 WMMA GEMM: C(f32, MxN) = A(bf16, MxK) * Bt(bf16, NxK transposed)
// One wave computes a 32x64 tile; K is a compile-time constant so the k-loop
// fully unrolls and loads pipeline above the WMMAs.
// ---------------------------------------------------------------------------
template <int K>
__global__ __launch_bounds__(32) void gemm_bf16_kernel(
    const __bf16* __restrict__ A, const __bf16* __restrict__ Bt,
    float* __restrict__ C, int M, int N) {
    int lane = threadIdx.x;
    int n0 = blockIdx.x * 64;
    int m0 = blockIdx.y * 32;
    v8f acc[2][4] = {};
#pragma unroll
    for (int k0 = 0; k0 < K; k0 += 32) {
        v16bf a0 = load_a_frag(A, K, m0,      k0, lane);
        v16bf a1 = load_a_frag(A, K, m0 + 16, k0, lane);
#pragma unroll
        for (int j = 0; j < 4; ++j) {
            v16bf b = load_bt_frag(Bt, K, k0, n0 + j * 16, lane);
            acc[0][j] = wmma_bf16(a0, b, acc[0][j]);
            acc[1][j] = wmma_bf16(a1, b, acc[1][j]);
        }
    }
#pragma unroll
    for (int half = 0; half < 2; ++half) {
        int mb = m0 + half * 16 + ((lane >> 4) << 3);
#pragma unroll
        for (int j = 0; j < 4; ++j) {
            int n = n0 + j * 16 + (lane & 15);
#pragma unroll
            for (int e = 0; e < 8; ++e)
                C[(long)(mb + e) * N + n] = acc[half][j][e];
        }
    }
}

// ---------------------------------------------------------------------------
// t = a + b; out = t * rsqrt(mean(t^2) + eps) * scale   (row = 256 elems)
// ---------------------------------------------------------------------------
__global__ __launch_bounds__(256) void add_rmsnorm_kernel(
    const float* __restrict__ a, const float* __restrict__ b,
    const float* __restrict__ scale, float* __restrict__ out_f32,
    __bf16* __restrict__ out_bf) {
    __shared__ float red[8];
    int row = blockIdx.x;
    int e   = threadIdx.x;
    float t  = a[(long)row * EMB + e] + b[(long)row * EMB + e];
    float ss = t * t;
#pragma unroll
    for (int off = 16; off > 0; off >>= 1) ss += __shfl_xor(ss, off);
    int wave = e >> 5, lane = e & 31;
    if (lane == 0) red[wave] = ss;
    __syncthreads();
    if (wave == 0) {
        float v2 = (lane < 8) ? red[lane] : 0.f;
#pragma unroll
        for (int off = 4; off > 0; off >>= 1) v2 += __shfl_xor(v2, off);
        if (lane == 0) red[0] = v2;
    }
    __syncthreads();
    float r = t * rsqrtf(red[0] * (1.f / 256.f) + 1e-6f) * scale[e];
    if (out_f32) out_f32[(long)row * EMB + e] = r;
    if (out_bf)  out_bf[(long)row * EMB + e] = (__bf16)r;
}

// ---------------------------------------------------------------------------
// SwiGLU: m = swish(gate) * lin ; gl holds [gate|lin] per row (512 wide)
// ---------------------------------------------------------------------------
__global__ __launch_bounds__(256) void swiglu_kernel(
    const float* __restrict__ gl, __bf16* __restrict__ mbf) {
    long i = (long)blockIdx.x * 256 + threadIdx.x;
    long row = i >> 8;
    int  e   = (int)(i & 255);
    float gate = gl[row * 512 + e];
    float lin  = gl[row * 512 + 256 + e];
    mbf[i] = (__bf16)(swishf(gate) * lin);
}

// ---------------------------------------------------------------------------
// Host-side orchestration
// ---------------------------------------------------------------------------
extern "C" void kernel_launch(void* const* d_in, const int* in_sizes, int n_in,
                              void* d_out, int out_size, void* d_ws, size_t ws_size,
                              hipStream_t stream) {
    const float* x        = (const float*)d_in[0];
    const float* wq       = (const float*)d_in[1];
    const float* wk       = (const float*)d_in[2];
    const float* wv       = (const float*)d_in[3];
    const float* wg       = (const float*)d_in[4];
    const float* wo       = (const float*)d_in[5];
    const float* gn_scale = (const float*)d_in[6];
    const float* gn_bias  = (const float*)d_in[7];
    const float* ln1      = (const float*)d_in[8];
    const float* ln2      = (const float*)d_in[9];
    const float* w_gate   = (const float*)d_in[10];
    const float* w_lin    = (const float*)d_in[11];
    const float* w_out    = (const float*)d_in[12];
    float* out = (float*)d_out;

    char* ws = (char*)d_ws;
    size_t off = 0;
    auto take = [&](size_t bytes) -> char* {
        char* p = ws + off;
        off += (bytes + 255) & ~(size_t)255;
        return p;
    };
    const size_t NROW = (size_t)ROWS;
    __bf16* xbf     = (__bf16*)take(NROW * EMB * 2);
    __bf16* wqkvgT  = (__bf16*)take(1024 * 256 * 2);
    __bf16* woT     = (__bf16*)take(256 * 256 * 2);
    __bf16* wglT    = (__bf16*)take(512 * 256 * 2);
    __bf16* woutT   = (__bf16*)take(256 * 256 * 2);
    __bf16* qbf     = (__bf16*)take(NROW * EMB * 2);
    __bf16* kbf     = (__bf16*)take(NROW * EMB * 2);
    __bf16* vbf     = (__bf16*)take(NROW * EMB * 2);
    float*  gbuf    = (float*)take(NROW * EMB * 4);
    float*  ybuf    = (float*)take(NROW * EMB * 4);
    __bf16* zbf     = (__bf16*)take(NROW * EMB * 2);
    float*  retbuf  = (float*)take(NROW * EMB * 4);   // reused for ffn
    float*  h1buf   = (float*)take(NROW * EMB * 4);
    __bf16* h1bf    = (__bf16*)take(NROW * EMB * 2);
    float*  glbuf   = (float*)take(NROW * 512 * 4);
    __bf16* mbf     = (__bf16*)take(NROW * EMB * 2);
    (void)ws_size; (void)n_in; (void)in_sizes; (void)out_size;

    // 1. convert x and weights (weights transposed) to bf16
    convert_x_kernel<<<(ROWS * EMB) / 256, 256, 0, stream>>>(x, xbf, ROWS * EMB);
    convert_w_kernel<<<(2048 * 256) / 256, 256, 0, stream>>>(
        wq, wk, wv, wg, wo, w_gate, w_lin, w_out, wqkvgT, woT, wglT, woutT);

    // 2. fused q/k/v/gate projection (M=8192, N=1024, K=256)
    proj_kernel<<<dim3(1024 / 64, ROWS / 32), 32, 0, stream>>>(
        xbf, wqkvgT, qbf, kbf, vbf, gbuf);

    // 3. streaming retention with decay
    retention_kernel<<<dim3(BATCH * NHEAD, SEQ / 128), 256, 0, stream>>>(
        qbf, kbf, vbf, ybuf);

    // 4. per-head GroupNorm + swish gate -> z (bf16)
    gn_gate_kernel<<<ROWS, 256, 0, stream>>>(ybuf, gbuf, gn_scale, gn_bias, zbf);

    // 5. ret = z @ wo
    gemm_bf16_kernel<EMB><<<dim3(EMB / 64, ROWS / 32), 32, 0, stream>>>(
        zbf, woT, retbuf, ROWS, EMB);

    // 6. h1 = rmsnorm(x + ret, ln1)
    add_rmsnorm_kernel<<<ROWS, 256, 0, stream>>>(x, retbuf, ln1, h1buf, h1bf);

    // 7. [gate|lin] = h1 @ [w_gate|w_lin]  (N=512)
    gemm_bf16_kernel<EMB><<<dim3(512 / 64, ROWS / 32), 32, 0, stream>>>(
        h1bf, wglT, glbuf, ROWS, 512);

    // 8. m = swish(gate) * lin
    swiglu_kernel<<<(ROWS * EMB) / 256, 256, 0, stream>>>(glbuf, mbf);

    // 9. ffn = m @ w_out (reuse ret buffer)
    gemm_bf16_kernel<EMB><<<dim3(EMB / 64, ROWS / 32), 32, 0, stream>>>(
        mbf, woutT, retbuf, ROWS, EMB);

    // 10. out = rmsnorm(h1 + ffn, ln2)
    add_rmsnorm_kernel<<<ROWS, 256, 0, stream>>>(h1buf, retbuf, ln2, out, nullptr);
}
